// Model_43791486550051
// MI455X (gfx1250) — matrix operand order repaired and checked by reading, not compiled, not run on hardware
//
#include <hip/hip_runtime.h>
#include <hip/hip_bf16.h>

// ---- problem constants ----
constexpr int kH   = 128;     // hidden
constexpr int kW2V = 250;     // w2v size
constexpr int kV   = 71347;   // vocab
constexpr int kB   = 20;      // batch
constexpr int kS   = 50;      // encoder seq len
constexpr int kL   = 40;      // decoder steps
constexpr int kM   = kL * kB; // 800 rows of the big GEMM
constexpr int kG   = 3 * kH;  // 384 gate width

typedef __attribute__((ext_vector_type(16))) _Float16 v16h;
typedef __attribute__((ext_vector_type(8)))  _Float16 half8;
typedef __attribute__((ext_vector_type(8)))  float    v8f;

__device__ __forceinline__ float sigm(float x) { return 1.0f / (1.0f + __expf(-x)); }

// ---------------------------------------------------------------------------
// 1) out_W (V,H) f32 -> f16 (same layout: row n holds K=0..127 contiguously,
//    which is exactly the per-lane contiguous-K chunk the WMMA B operand wants)
// ---------------------------------------------------------------------------
__global__ void convert_w_kernel(const float* __restrict__ outW,
                                 _Float16* __restrict__ Wh, int n) {
  for (int i = blockIdx.x * blockDim.x + threadIdx.x; i < n;
       i += gridDim.x * blockDim.x)
    Wh[i] = (_Float16)outW[i];
}

// ---------------------------------------------------------------------------
// 2) encoder embedding: emb[s,b,h] = x[b,s,:] . enc_emb_W[h,:] + b[h]
// ---------------------------------------------------------------------------
__global__ __launch_bounds__(128)
void emb_kernel(const float* __restrict__ x, const float* __restrict__ W,
                const float* __restrict__ bias, float* __restrict__ emb) {
  const int blk = blockIdx.x;          // blk = s*kB + b
  const int s = blk / kB, b = blk % kB;
  __shared__ float xr[kW2V];
  const float* xp = x + ((size_t)b * kS + s) * kW2V;
  if (threadIdx.x < 125) {
    xr[threadIdx.x]       = xp[threadIdx.x];
    xr[threadIdx.x + 125] = xp[threadIdx.x + 125];
  }
  __syncthreads();
  const int h = threadIdx.x;
  float sum = bias[h];
  const float* w = W + (size_t)h * kW2V;
  #pragma unroll 5
  for (int k = 0; k < kW2V; ++k) sum += xr[k] * w[k];
  emb[(size_t)blk * kH + h] = sum;
}

// ---------------------------------------------------------------------------
// 3) input-gate precompute (both directions): gi[s,b,j] = emb[s,b,:].Wih[j,:] + bih[j]
// ---------------------------------------------------------------------------
__global__ __launch_bounds__(384)
void enc_gi_kernel(const float* __restrict__ emb,
                   const float* __restrict__ WihF, const float* __restrict__ bihF,
                   const float* __restrict__ WihB, const float* __restrict__ bihB,
                   float* __restrict__ giF, float* __restrict__ giB) {
  const int sb = blockIdx.x, dir = blockIdx.y;
  const float* Wih = dir ? WihB : WihF;
  const float* bih = dir ? bihB : bihF;
  float* gi = dir ? giB : giF;
  __shared__ float er[kH];
  if (threadIdx.x < kH) er[threadIdx.x] = emb[(size_t)sb * kH + threadIdx.x];
  __syncthreads();
  const int j = threadIdx.x;
  float sum = bih[j];
  const float* w = Wih + (size_t)j * kH;
  #pragma unroll 8
  for (int k = 0; k < kH; ++k) sum += er[k] * w[k];
  gi[(size_t)sb * kG + j] = sum;
}

// ---------------------------------------------------------------------------
// 4) serial encoder scans, one block per direction. Only gh = h.Whh^T stays
//    in the loop; gi comes precomputed from global.
// ---------------------------------------------------------------------------
__global__ __launch_bounds__(1024)
void enc_scan_kernel(const float* __restrict__ giF, const float* __restrict__ giB,
                     const float* __restrict__ WhhF, const float* __restrict__ bhhF,
                     const float* __restrict__ WhhB, const float* __restrict__ bhhB,
                     float* __restrict__ outsF, float* __restrict__ outsBwd) {
  const int dir = blockIdx.x;
  const float* gi  = dir ? giB  : giF;
  const float* Whh = dir ? WhhB : WhhF;
  const float* bhh = dir ? bhhB : bhhF;
  float* outs = dir ? outsBwd : outsF;
  __shared__ float hL[kB * kH];    // 10 KB
  __shared__ float ghL[kB * kG];   // 30 KB
  const int tid = threadIdx.x;
  for (int i = tid; i < kB * kH; i += 1024) hL[i] = 0.0f;
  __syncthreads();
  for (int step = 0; step < kS; ++step) {
    const int s = dir ? (kS - 1 - step) : step;
    for (int idx = tid; idx < kB * kG; idx += 1024) {
      const int b = idx / kG, j = idx % kG;
      const float* hr = &hL[b * kH];
      const float* wr = Whh + (size_t)j * kH;
      float sum = bhh[j];
      #pragma unroll 8
      for (int k = 0; k < kH; ++k) sum += hr[k] * wr[k];
      ghL[idx] = sum;
    }
    __syncthreads();
    const float* gis = gi + (size_t)s * kB * kG;
    for (int idx = tid; idx < kB * kH; idx += 1024) {
      const int b = idx >> 7, k = idx & 127;
      const float ir  = gis[b * kG + k];
      const float iz  = gis[b * kG + kH + k];
      const float inn = gis[b * kG + 2 * kH + k];
      const float hr  = ghL[b * kG + k];
      const float hz  = ghL[b * kG + kH + k];
      const float hn  = ghL[b * kG + 2 * kH + k];
      const float r = sigm(ir + hr);
      const float z = sigm(iz + hz);
      const float n = tanhf(inn + r * hn);
      const float hnew = (1.0f - z) * n + z * hL[idx];
      hL[idx] = hnew;
      outs[(size_t)s * kB * kH + idx] = hnew;   // bwd stored at original s
    }
    __syncthreads();
  }
}

// ---------------------------------------------------------------------------
// 5) enc_out = outs_f + outs_b[::-1]  (bwd already stored at original s)
// ---------------------------------------------------------------------------
__global__ void add_kernel(const float* __restrict__ a, const float* __restrict__ b,
                           float* __restrict__ c, int n) {
  const int i = blockIdx.x * blockDim.x + threadIdx.x;
  if (i < n) c[i] = a[i] + b[i];
}

// ---------------------------------------------------------------------------
// 6) e0[b,h] = randn[b,:] . dec_emb_W[h,:]   (dense, memory-bound reduction)
// ---------------------------------------------------------------------------
__global__ __launch_bounds__(256)
void dec_e0_kernel(const float* __restrict__ randn, const float* __restrict__ dembW,
                   float* __restrict__ e0) {
  const int h = blockIdx.x, b = blockIdx.y;
  const float* r = randn + (size_t)b * kV;
  const float* w = dembW + (size_t)h * kV;
  float s = 0.0f;
  for (int v = threadIdx.x; v < kV; v += 256) s += r[v] * w[v];
  __shared__ float red[256];
  red[threadIdx.x] = s; __syncthreads();
  for (int st = 128; st > 0; st >>= 1) {
    if (threadIdx.x < st) red[threadIdx.x] += red[threadIdx.x + st];
    __syncthreads();
  }
  if (threadIdx.x == 0) e0[b * kH + h] = red[0];
}

// ---------------------------------------------------------------------------
// 7) decoder input gates for ALL steps in parallel (teacher forcing: input at
//    step t>=1 is onehot(y[:,t-1]) -> embedding is a column gather)
// ---------------------------------------------------------------------------
__global__ __launch_bounds__(384)
void dec_gi_kernel(const float* __restrict__ e0, const float* __restrict__ dembW,
                   const float* __restrict__ dembB, const int* __restrict__ y,
                   const float* __restrict__ dWih, const float* __restrict__ dbih,
                   float* __restrict__ giDec) {
  const int m = blockIdx.x;            // m = t*kB + b
  const int t = m / kB, b = m % kB;
  __shared__ float er[kH];
  const int k = threadIdx.x;
  if (k < kH) {
    float v;
    if (t == 0) v = e0[b * kH + k];
    else {
      const int tok = y[b * kL + (t - 1)];
      v = dembW[(size_t)k * kV + tok];
    }
    er[k] = v + dembB[k];
  }
  __syncthreads();
  const int j = threadIdx.x;
  float sum = dbih[j];
  const float* w = dWih + (size_t)j * kH;
  #pragma unroll 8
  for (int kk = 0; kk < kH; ++kk) sum += er[kk] * w[kk];
  giDec[(size_t)m * kG + j] = sum;
}

// ---------------------------------------------------------------------------
// 8) serial decoder hidden chain; h0 = forward encoder final hidden; stores
//    every h_t (hs[t] feeds output t)
// ---------------------------------------------------------------------------
__global__ __launch_bounds__(1024)
void dec_scan_kernel(const float* __restrict__ giDec, const float* __restrict__ dWhh,
                     const float* __restrict__ dbhh, const float* __restrict__ outsF,
                     float* __restrict__ hs) {
  __shared__ float hL[kB * kH];
  __shared__ float ghL[kB * kG];
  const int tid = threadIdx.x;
  for (int i = tid; i < kB * kH; i += 1024)
    hL[i] = outsF[(size_t)(kS - 1) * kB * kH + i];   // hf
  __syncthreads();
  for (int t = 0; t < kL; ++t) {
    for (int idx = tid; idx < kB * kG; idx += 1024) {
      const int b = idx / kG, j = idx % kG;
      const float* hr = &hL[b * kH];
      const float* wr = dWhh + (size_t)j * kH;
      float sum = dbhh[j];
      #pragma unroll 8
      for (int k = 0; k < kH; ++k) sum += hr[k] * wr[k];
      ghL[idx] = sum;
    }
    __syncthreads();
    const float* gis = giDec + (size_t)t * kB * kG;
    for (int idx = tid; idx < kB * kH; idx += 1024) {
      const int b = idx >> 7, k = idx & 127;
      const float ir  = gis[b * kG + k];
      const float iz  = gis[b * kG + kH + k];
      const float inn = gis[b * kG + 2 * kH + k];
      const float hr  = ghL[b * kG + k];
      const float hz  = ghL[b * kG + kH + k];
      const float hn  = ghL[b * kG + 2 * kH + k];
      const float r = sigm(ir + hr);
      const float z = sigm(iz + hz);
      const float n = tanhf(inn + r * hn);
      const float hnew = (1.0f - z) * n + z * hL[idx];
      hL[idx] = hnew;
      hs[(size_t)t * kB * kH + idx] = hnew;
    }
    __syncthreads();
  }
}

// ---------------------------------------------------------------------------
// 9) attention + concat + tanh for all 800 (t,b) pairs in parallel; emits the
//    f16 A-matrix (row-major (kM, kH)) for the big WMMA GEMM.
// ---------------------------------------------------------------------------
__global__ __launch_bounds__(128)
void attn_kernel(const float* __restrict__ hs, const float* __restrict__ encOut,
                 const float* __restrict__ catW, const float* __restrict__ catB,
                 _Float16* __restrict__ Ah) {
  const int m = blockIdx.x;            // m = t*kB + b
  const int t = m / kB, b = m % kB;
  __shared__ float h2[kH];
  __shared__ float sc[64];
  __shared__ float cin[2 * kH];
  const int tid = threadIdx.x;
  h2[tid] = hs[((size_t)t * kB + b) * kH + tid];
  __syncthreads();
  if (tid < kS) {
    const float* eo = encOut + ((size_t)tid * kB + b) * kH;
    float s = 0.0f;
    #pragma unroll 8
    for (int k = 0; k < kH; ++k) s += h2[k] * eo[k];
    sc[tid] = s;
  }
  __syncthreads();
  if (tid == 0) {
    float mx = sc[0];
    for (int s = 1; s < kS; ++s) mx = fmaxf(mx, sc[s]);
    float sum = 0.0f;
    for (int s = 0; s < kS; ++s) { const float e = __expf(sc[s] - mx); sc[s] = e; sum += e; }
    const float inv = 1.0f / sum;
    for (int s = 0; s < kS; ++s) sc[s] *= inv;
  }
  __syncthreads();
  {
    float c = 0.0f;
    for (int s = 0; s < kS; ++s) c += sc[s] * encOut[((size_t)s * kB + b) * kH + tid];
    cin[tid] = h2[tid];
    cin[kH + tid] = c;
  }
  __syncthreads();
  float acc = catB[tid];
  const float* w = catW + (size_t)tid * (2 * kH);
  #pragma unroll 8
  for (int i = 0; i < 2 * kH; ++i) acc += cin[i] * w[i];
  Ah[(size_t)m * kH + tid] = (_Float16)tanhf(acc);
}

// ---------------------------------------------------------------------------
// 10) big output GEMM via WMMA: (800,128)f16 x (128,71347)f16 -> f32 logits.
//     Per block: one 16-row A tile (4 KB, contiguous) is staged into LDS with
//     GLOBAL_LOAD_ASYNC_TO_LDS_B128 (ASYNCcnt) and shared by all 4 waves.
//     Per wave: 16(M) x 64(N) tile -> 16 x v_wmma_f32_16x16x32_f16 (K=128).
//     A-operand lane layout (16-bit A 16x32): lane l (l16=l&15,
//     half=l>>4) holds K = k0+16*half+[0..15] of row l16 (matches B).
//     B-operand (32x16): lane holds 16 contiguous K (k0+16*half..+15) of
//     column n0+l16 -> one 32-byte contiguous load from Wh (V,128).
// ---------------------------------------------------------------------------
__global__ __launch_bounds__(128)
void out_gemm_kernel(const _Float16* __restrict__ Ah, const _Float16* __restrict__ Wh,
                     const float* __restrict__ outB, float* __restrict__ out) {
  __shared__ __align__(32) _Float16 As[16 * kH];   // 4 KB A tile
  const int tid  = threadIdx.x;
  const int wave = tid >> 5;
  const int lane = tid & 31;
  const int halfw = lane >> 4;
  const int l16  = lane & 15;
  const int m0 = blockIdx.y * 16;

  // ---- async-stage the A tile: 256 x 16B chunks, 2 per thread ----
  {
    const _Float16* g0 = Ah + (size_t)m0 * kH + tid * 8;      // 16 B per chunk
    unsigned long long ga0 = (unsigned long long)(uintptr_t)g0;
    unsigned long long ga1 = ga0 + 2048ull;                    // second half
    unsigned la0 = (unsigned)(uintptr_t)(&As[tid * 8]);        // LDS addr = low 32b
    unsigned la1 = la0 + 2048u;
    asm volatile("global_load_async_to_lds_b128 %0, %1, off" :: "v"(la0), "v"(ga0) : "memory");
    asm volatile("global_load_async_to_lds_b128 %0, %1, off" :: "v"(la1), "v"(ga1) : "memory");
    asm volatile("s_wait_asynccnt 0x0" ::: "memory");
  }
  __syncthreads();

  const int nwave0 = blockIdx.x * 256 + wave * 64;   // 4 N-subtiles per wave
  v8f acc[4] = {};
  const _Float16* arow = As + l16 * kH + halfw * 16;
  #pragma unroll
  for (int k0 = 0; k0 < kH; k0 += 32) {
    union { v16h v; half8 h[2]; } a;
    a.h[0] = *(const half8*)(arow + k0);
    a.h[1] = *(const half8*)(arow + k0 + 8);
    #pragma unroll
    for (int nt = 0; nt < 4; ++nt) {
      int nb = nwave0 + nt * 16 + l16;        // clamp loads; guard stores
      if (nb >= kV) nb = kV - 1;
      const v16h bmat = *(const v16h*)(Wh + (size_t)nb * kH + k0 + halfw * 16);
      acc[nt] = __builtin_amdgcn_wmma_f32_16x16x32_f16(
          /*neg_a=*/false, a.v, /*neg_b=*/false, bmat,
          /*c_mod=*/(short)0, acc[nt], /*reuse_a=*/false, /*reuse_b=*/false);
    }
  }

  // hoist the m -> (b,t) row mapping out of the N loop (kills div/mod VALU)
  size_t rowOff[8];
  {
    const int mbase = m0 + halfw * 8;          // C layout: lanes>=16 hold M=i+8
    #pragma unroll
    for (int i = 0; i < 8; ++i) {
      const int m = mbase + i;                 // m = t*kB + b
      const int t = m / kB, bb = m % kB;
      rowOff[i] = (size_t)(bb * kL + t) * kV;  // (B,L,V)
    }
  }
  #pragma unroll
  for (int nt = 0; nt < 4; ++nt) {
    const int nn = nwave0 + nt * 16 + l16;
    if (nn < kV) {
      const float bias = outB[nn];
      #pragma unroll
      for (int i = 0; i < 8; ++i) out[rowOff[i] + nn] = acc[nt][i] + bias;
    }
  }
}

// ---------------------------------------------------------------------------
// 11) in-place softmax over each length-V row of d_out (800 rows)
// ---------------------------------------------------------------------------
__global__ __launch_bounds__(256)
void softmax_kernel(float* __restrict__ out) {
  float* p = out + (size_t)blockIdx.x * kV;
  __shared__ float red[256];
  const int tid = threadIdx.x;
  float mx = -1e30f;
  for (int v = tid; v < kV; v += 256) mx = fmaxf(mx, p[v]);
  red[tid] = mx; __syncthreads();
  for (int st = 128; st > 0; st >>= 1) {
    if (tid < st) red[tid] = fmaxf(red[tid], red[tid + st]);
    __syncthreads();
  }
  mx = red[0]; __syncthreads();
  float s = 0.0f;
  for (int v = tid; v < kV; v += 256) s += __expf(p[v] - mx);
  red[tid] = s; __syncthreads();
  for (int st = 128; st > 0; st >>= 1) {
    if (tid < st) red[tid] += red[tid + st];
    __syncthreads();
  }
  const float inv = 1.0f / red[0];
  for (int v = tid; v < kV; v += 256) p[v] = __expf(p[v] - mx) * inv;
}

// ---------------------------------------------------------------------------
extern "C" void kernel_launch(void* const* d_in, const int* in_sizes, int n_in,
                              void* d_out, int out_size, void* d_ws, size_t ws_size,
                              hipStream_t stream) {
  const float* x     = (const float*)d_in[0];
  const int*   y     = (const int*)  d_in[2];
  const float* randn = (const float*)d_in[3];
  const float* eembW = (const float*)d_in[4];
  const float* eembB = (const float*)d_in[5];
  const float* WihF  = (const float*)d_in[6];
  const float* WhhF  = (const float*)d_in[7];
  const float* bihF  = (const float*)d_in[8];
  const float* bhhF  = (const float*)d_in[9];
  const float* WihB  = (const float*)d_in[10];
  const float* WhhB  = (const float*)d_in[11];
  const float* bihB  = (const float*)d_in[12];
  const float* bhhB  = (const float*)d_in[13];
  const float* dembW = (const float*)d_in[14];
  const float* dembB = (const float*)d_in[15];
  const float* dWih  = (const float*)d_in[16];
  const float* dWhh  = (const float*)d_in[17];
  const float* dbih  = (const float*)d_in[18];
  const float* dbhh  = (const float*)d_in[19];
  const float* catW  = (const float*)d_in[20];
  const float* catB  = (const float*)d_in[21];
  const float* outW  = (const float*)d_in[22];
  const float* outB  = (const float*)d_in[23];
  float* out = (float*)d_out;

  char* base = (char*)d_ws;
  size_t off = 0;
  auto take = [&](size_t bytes) -> void* {
    void* p = base + off;
    off = (off + bytes + 255) & ~(size_t)255;
    return p;
  };
  float*    emb     = (float*)take((size_t)kS * kB * kH * 4);
  float*    giF     = (float*)take((size_t)kS * kB * kG * 4);
  float*    giB     = (float*)take((size_t)kS * kB * kG * 4);
  float*    outsF   = (float*)take((size_t)kS * kB * kH * 4);
  float*    outsBwd = (float*)take((size_t)kS * kB * kH * 4);
  float*    encOut  = (float*)take((size_t)kS * kB * kH * 4);
  float*    e0      = (float*)take((size_t)kB * kH * 4);
  float*    giDec   = (float*)take((size_t)kM * kG * 4);
  float*    hs      = (float*)take((size_t)kL * kB * kH * 4);
  _Float16* Ah      = (_Float16*)take((size_t)kM * kH * 2);
  _Float16* Wh      = (_Float16*)take(((size_t)kV * kH + 256) * 2);

  // weight conversion + encoder front-end (all parallel)
  convert_w_kernel<<<4096, 256, 0, stream>>>(outW, Wh, kV * kH);
  emb_kernel<<<kS * kB, kH, 0, stream>>>(x, eembW, eembB, emb);
  enc_gi_kernel<<<dim3(kS * kB, 2), kG, 0, stream>>>(emb, WihF, bihF, WihB, bihB,
                                                     giF, giB);
  // serial encoder scans (fwd + bwd in parallel blocks)
  enc_scan_kernel<<<2, 1024, 0, stream>>>(giF, giB, WhhF, bhhF, WhhB, bhhB,
                                          outsF, outsBwd);
  add_kernel<<<(kS * kB * kH + 255) / 256, 256, 0, stream>>>(outsF, outsBwd,
                                                             encOut, kS * kB * kH);
  // decoder front-end (parallel)
  dec_e0_kernel<<<dim3(kH, kB), 256, 0, stream>>>(randn, dembW, e0);
  dec_gi_kernel<<<kM, kG, 0, stream>>>(e0, dembW, dembB, y, dWih, dbih, giDec);
  // serial decoder hidden chain
  dec_scan_kernel<<<1, 1024, 0, stream>>>(giDec, dWhh, dbhh, outsF, hs);
  // attention/concat for all steps (parallel) -> f16 A matrix
  attn_kernel<<<kM, kH, 0, stream>>>(hs, encOut, catW, catB, Ah);
  // big WMMA GEMM: block covers 16M x 256N (4 waves x 4 subtiles); 50 M-tiles
  out_gemm_kernel<<<dim3((kV + 255) / 256, kM / 16), 128, 0, stream>>>(Ah, Wh, outB, out);
  // per-row softmax over V
  softmax_kernel<<<kM, 256, 0, stream>>>(out);
}
